// MultiHeadAttention_33646773797336
// MI455X (gfx1250) — matrix operand order repaired and checked by reading, not compiled, not run on hardware
//
#include <hip/hip_runtime.h>

// ---------------------------------------------------------------------------
// MHA for MI455X (gfx1250, wave32): bf16 WMMA pipeline
//   stage 0: one-shot fp32 -> bf16 conversion (query pre-scaled by 1/sqrt(dk))
//   stage 1: Q/K/V = xbf @ Wbf^T (bf16 GEMM, ping-pong register buffers)
//   stage 2: flash attention; K/V chunks staged block-wide into LDS via
//            global_load_async_to_lds_b128 (ASYNCcnt), double-buffered
//   stage 3: out = ctx @ Wo^T + bo (fp32 out)
// ---------------------------------------------------------------------------

typedef __attribute__((ext_vector_type(16))) __bf16 v16bf;
typedef __attribute__((ext_vector_type(8)))  float  v8f;

#define BATCH 2
#define SEQ   2048
#define DIM   1024
#define NHEAD 16
#define DK    64

static __device__ inline __bf16 f2bf(float f) { return (__bf16)f; }

static __device__ inline v8f wmma_bf16(v16bf a, v16bf b, v8f c) {
    return __builtin_amdgcn_wmma_f32_16x16x32_bf16(
        false, a, false, b, (short)0, c, false, false);
}

static __device__ inline v8f zero8() {
    v8f z = {0.f, 0.f, 0.f, 0.f, 0.f, 0.f, 0.f, 0.f};
    return z;
}

// A operand (16x32): lane holds row r=lane%16, K chunks [k0,k0+8) and
// [k0+16,k0+24), k0=(lane/16)*8  -> two 16B contiguous loads
static __device__ inline v16bf load_a_bf(const __bf16* p, int ld, int lane) {
    int r = lane & 15, k0 = (lane >> 4) << 3;
    const __bf16* q = p + (size_t)r * ld + k0;
    v16bf a;
#pragma unroll
    for (int i = 0; i < 8; ++i) { a[i] = q[i]; a[i + 8] = q[16 + i]; }
    return a;
}

// B operand (32x16): lane holds column n=lane%16 (== source row n,
// K-contiguous), K chunks [k0,k0+8) and [k0+16,k0+24), k0=(lane/16)*8
static __device__ inline v16bf load_b_bf(const __bf16* p, int ld, int lane) {
    int n = lane & 15, k0 = (lane >> 4) << 3;
    const __bf16* q = p + (size_t)n * ld + k0;
    v16bf b;
#pragma unroll
    for (int i = 0; i < 8; ++i) { b[i] = q[i]; b[i + 8] = q[16 + i]; }
    return b;
}

// ---- CDNA5 async global->LDS copy (16B per lane, tracked by ASYNCcnt) ----
static __device__ inline void async_copy_b128(unsigned lds_byte_off, const void* g) {
    asm volatile("global_load_async_to_lds_b128 %0, %1, off"
                 :: "v"(lds_byte_off), "v"(g) : "memory");
}
static __device__ inline void wait_async_le2() {
    asm volatile("s_wait_asynccnt 0x2" ::: "memory");
}
static __device__ inline void wait_async_0() {
    asm volatile("s_wait_asynccnt 0x0" ::: "memory");
}

// ---------------------------------------------------------------------------
// Stage 0: bulk fp32 -> bf16 (8 elems/thread)
// ---------------------------------------------------------------------------
__global__ __launch_bounds__(256) void cvt_f32_bf16(
    const float* __restrict__ src, __bf16* __restrict__ dst, int n, float scale)
{
    int i = (blockIdx.x * 256 + threadIdx.x) * 8;
    if (i < n) {
#pragma unroll
        for (int j = 0; j < 8; ++j) dst[i + j] = f2bf(src[i + j] * scale);
    }
}

// ---------------------------------------------------------------------------
// Stage 1: fused QKV projection. grid = (M/128, N/64, 3); block = 256 (8 waves)
// Ping-pong K loop (two fixed register sets, no rotation copies).
// ---------------------------------------------------------------------------
__global__ __launch_bounds__(256) void mha_proj_qkv(
    const __bf16* __restrict__ Xq, const __bf16* __restrict__ Xk,
    const __bf16* __restrict__ Xv, const __bf16* __restrict__ Wq,
    const __bf16* __restrict__ Wk, const __bf16* __restrict__ Wv,
    __bf16* __restrict__ Qbf, __bf16* __restrict__ Kbf, __bf16* __restrict__ Vt)
{
    const int z     = blockIdx.z;
    const __bf16* X = (z == 0) ? Xq : (z == 1) ? Xk : Xv;
    const __bf16* W = (z == 0) ? Wq : (z == 1) ? Wk : Wv;

    const int lane = threadIdx.x & 31, wave = threadIdx.x >> 5;
    const int row  = blockIdx.x * 128 + wave * 16;
    const int colb = blockIdx.y * 64;

    const __bf16* Arow = X + (size_t)row * DIM;
    const __bf16* Wc[4];
#pragma unroll
    for (int t = 0; t < 4; ++t) Wc[t] = W + (size_t)(colb + t * 16) * DIM;

    v8f acc[4];
#pragma unroll
    for (int t = 0; t < 4; ++t) acc[t] = zero8();

    v16bf a0 = load_a_bf(Arow, DIM, lane);
    v16bf b0[4];
#pragma unroll
    for (int t = 0; t < 4; ++t) b0[t] = load_b_bf(Wc[t], DIM, lane);

    for (int kk = 0; kk < DIM; kk += 64) {
        // load set1 (k+32) -- overlaps wmma(set0)
        v16bf a1 = load_a_bf(Arow + kk + 32, DIM, lane);
        v16bf b1[4];
#pragma unroll
        for (int t = 0; t < 4; ++t) b1[t] = load_b_bf(Wc[t] + kk + 32, DIM, lane);

#pragma unroll
        for (int t = 0; t < 4; ++t) acc[t] = wmma_bf16(a0, b0[t], acc[t]);

        // load set0 (k+64, clamped branchlessly) -- overlaps wmma(set1)
        int kn = (kk + 64 < DIM) ? kk + 64 : 0;
        a0 = load_a_bf(Arow + kn, DIM, lane);
#pragma unroll
        for (int t = 0; t < 4; ++t) b0[t] = load_b_bf(Wc[t] + kn, DIM, lane);

#pragma unroll
        for (int t = 0; t < 4; ++t) acc[t] = wmma_bf16(a1, b1[t], acc[t]);
    }

    const int half = lane >> 4, nlo = lane & 15;
#pragma unroll
    for (int t = 0; t < 4; ++t) {
#pragma unroll
        for (int j = 0; j < 8; ++j) {
            int m = row + j + 8 * half;
            int n = colb + t * 16 + nlo;
            int bb = m >> 11, s = m & (SEQ - 1);
            int h = n >> 6, d = n & (DK - 1);
            __bf16 v = f2bf(acc[t][j]);
            if (z == 0)      Qbf[(((size_t)(bb * NHEAD + h) * SEQ + s) * DK) + d] = v;
            else if (z == 1) Kbf[(((size_t)(bb * NHEAD + h) * SEQ + s) * DK) + d] = v;
            else             Vt[(((size_t)(bb * NHEAD + h) * DK + d) * SEQ) + s] = v;
        }
    }
}

// ---------------------------------------------------------------------------
// Stage 2: flash attention. grid = (Sq/128, H, B); block = 256 (8 waves).
// K/V chunks (32 keys) staged into LDS once per block via async copies,
// double-buffered; each wave owns 16 query rows (Q resident in registers).
// Dynamic LDS layout (32 KB):
//   [0,    8K)  kbuf[2][32][64] bf16
//   [8K,  16K)  vbuf[2][64][32] bf16
//   [16K, 32K)  pbuf[8][16][32] f32   (per-wave P C->A relayout)
// ---------------------------------------------------------------------------
__global__ __launch_bounds__(256) void mha_attn(
    const __bf16* __restrict__ Qbf, const __bf16* __restrict__ Kbf,
    const __bf16* __restrict__ Vt, __bf16* __restrict__ Ctx)
{
    extern __shared__ char smem[];

    const int tid  = threadIdx.x;
    const int lane = tid & 31, wave = tid >> 5;
    const int half = lane >> 4, nlo = lane & 15;
    const int b = blockIdx.z, h = blockIdx.y;
    const int q0 = blockIdx.x * 128 + wave * 16;

    const __bf16* Qh = Qbf + (size_t)(b * NHEAD + h) * SEQ * DK;
    const __bf16* Kh = Kbf + (size_t)(b * NHEAD + h) * SEQ * DK;
    const __bf16* Vh = Vt  + (size_t)(b * NHEAD + h) * DK * SEQ;

    // Q tile (16 x 64) resident in registers: two 16x32 A operands
    v16bf qa0 = load_a_bf(Qh + (size_t)q0 * DK,      DK, lane);
    v16bf qa1 = load_a_bf(Qh + (size_t)q0 * DK + 32, DK, lane);

    float mi[8], li[8];
    v8f   ctx[4];
#pragma unroll
    for (int j = 0; j < 8; ++j) { mi[j] = -1e30f; li[j] = 0.f; }
#pragma unroll
    for (int t = 0; t < 4; ++t) ctx[t] = zero8();

    // per-thread async copy addresses: 16B per lane, 256 threads = 4KB/chunk
    auto issue_copy = [&](int buf, int kc) {
        unsigned koff = (unsigned)(buf * 4096 + tid * 16);
        const __bf16* kg = Kh + (size_t)(kc + (tid >> 3)) * DK + ((tid & 7) << 3);
        async_copy_b128(koff, kg);
        unsigned voff = (unsigned)(8192 + buf * 4096 + tid * 16);
        const __bf16* vg = Vh + (size_t)(tid >> 2) * SEQ + kc + ((tid & 3) << 3);
        async_copy_b128(voff, vg);
    };

    issue_copy(0, 0);   // prologue: fill buffer 0

    for (int kc = 0; kc < SEQ; kc += 32) {
        const int buf = (kc >> 5) & 1;
        if (kc + 32 < SEQ) {
            issue_copy(buf ^ 1, kc + 32);   // overlap next copy with compute
            wait_async_le2();               // this wave's current-buf copies done
        } else {
            wait_async_0();
        }
        __syncthreads();                    // all waves' copies visible

        const __bf16* kb = (const __bf16*)(smem + buf * 4096);          // [32][64]
        const __bf16* vb = (const __bf16*)(smem + 8192 + buf * 4096);   // [64][32]
        float*        pb = (float*)(smem + 16384) + wave * 512;         // [16][32]

        // ---- scores: S(16x32) = Q(16x64) @ K_chunk^T (operands from LDS) ----
        v8f s0 = zero8(), s1 = zero8();
        s0 = wmma_bf16(qa0, load_b_bf(kb,                64, lane), s0);
        s1 = wmma_bf16(qa0, load_b_bf(kb + 16 * 64,      64, lane), s1);
        s0 = wmma_bf16(qa1, load_b_bf(kb + 32,           64, lane), s0);
        s1 = wmma_bf16(qa1, load_b_bf(kb + 16 * 64 + 32, 64, lane), s1);

        // ---- online softmax: row reductions across a 16-lane half ----
        float rowm[8];
#pragma unroll
        for (int j = 0; j < 8; ++j) rowm[j] = fmaxf(s0[j], s1[j]);
#pragma unroll
        for (int mask = 1; mask <= 8; mask <<= 1)
#pragma unroll
            for (int j = 0; j < 8; ++j)
                rowm[j] = fmaxf(rowm[j], __shfl_xor(rowm[j], mask, 32));

        float alpha[8];
#pragma unroll
        for (int j = 0; j < 8; ++j) {
            float mn = fmaxf(mi[j], rowm[j]);
            alpha[j] = __expf(mi[j] - mn);
            mi[j] = mn;
        }
#pragma unroll
        for (int j = 0; j < 8; ++j) {
            s0[j] = __expf(s0[j] - mi[j]);
            s1[j] = __expf(s1[j] - mi[j]);
        }
        float rs[8];
#pragma unroll
        for (int j = 0; j < 8; ++j) rs[j] = s0[j] + s1[j];
#pragma unroll
        for (int mask = 1; mask <= 8; mask <<= 1)
#pragma unroll
            for (int j = 0; j < 8; ++j) rs[j] += __shfl_xor(rs[j], mask, 32);
#pragma unroll
        for (int j = 0; j < 8; ++j) li[j] = li[j] * alpha[j] + rs[j];

#pragma unroll
        for (int t = 0; t < 4; ++t)
#pragma unroll
            for (int j = 0; j < 8; ++j) ctx[t][j] *= alpha[j];

        // ---- P: C-layout -> A-layout via per-wave LDS tile ----
#pragma unroll
        for (int j = 0; j < 8; ++j) {
            pb[(j + 8 * half) * 32 + nlo]      = s0[j];
            pb[(j + 8 * half) * 32 + nlo + 16] = s1[j];
        }
        __builtin_amdgcn_wave_barrier();
        v16bf pa;
        {
            int r = lane & 15, k0 = (lane >> 4) << 3;
#pragma unroll
            for (int i = 0; i < 8; ++i) {
                pa[i]     = f2bf(pb[r * 32 + k0 + i]);
                pa[i + 8] = f2bf(pb[r * 32 + k0 + 16 + i]);
            }
        }
        __builtin_amdgcn_wave_barrier();

        // ---- ctx += P(16x32) @ V_chunk(32x64) (V columns key-contiguous) ----
#pragma unroll
        for (int t = 0; t < 4; ++t) {
            v16bf vv = load_b_bf(vb + (t * 16) * 32, 32, lane);
            ctx[t] = wmma_bf16(pa, vv, ctx[t]);
        }

        __syncthreads();    // all waves done reading before buffer reuse
    }

    // ---- normalize and store concat layout [B, Sq, H*dk] as bf16 ----
#pragma unroll
    for (int t = 0; t < 4; ++t) {
#pragma unroll
        for (int j = 0; j < 8; ++j) {
            float v = ctx[t][j] / li[j];
            int s   = q0 + j + 8 * half;
            int col = h * DK + t * 16 + nlo;
            Ctx[((size_t)(b * SEQ + s)) * DIM + col] = f2bf(v);
        }
    }
}

// ---------------------------------------------------------------------------
// Stage 3: out = ctx @ Wo^T + bo. grid = (M/128, N/64); block = 256.
// Ping-pong K loop identical to stage 1.
// ---------------------------------------------------------------------------
__global__ __launch_bounds__(256) void mha_oproj(
    const __bf16* __restrict__ Ctx, const __bf16* __restrict__ Wo,
    const float* __restrict__ bo, float* __restrict__ Out)
{
    const int lane = threadIdx.x & 31, wave = threadIdx.x >> 5;
    const int row  = blockIdx.x * 128 + wave * 16;
    const int colb = blockIdx.y * 64;

    const __bf16* Arow = Ctx + (size_t)row * DIM;
    const __bf16* Wc[4];
#pragma unroll
    for (int t = 0; t < 4; ++t) Wc[t] = Wo + (size_t)(colb + t * 16) * DIM;

    v8f acc[4];
#pragma unroll
    for (int t = 0; t < 4; ++t) acc[t] = zero8();

    v16bf a0 = load_a_bf(Arow, DIM, lane);
    v16bf b0[4];
#pragma unroll
    for (int t = 0; t < 4; ++t) b0[t] = load_b_bf(Wc[t], DIM, lane);

    for (int kk = 0; kk < DIM; kk += 64) {
        v16bf a1 = load_a_bf(Arow + kk + 32, DIM, lane);
        v16bf b1[4];
#pragma unroll
        for (int t = 0; t < 4; ++t) b1[t] = load_b_bf(Wc[t] + kk + 32, DIM, lane);

#pragma unroll
        for (int t = 0; t < 4; ++t) acc[t] = wmma_bf16(a0, b0[t], acc[t]);

        int kn = (kk + 64 < DIM) ? kk + 64 : 0;
        a0 = load_a_bf(Arow + kn, DIM, lane);
#pragma unroll
        for (int t = 0; t < 4; ++t) b0[t] = load_b_bf(Wc[t] + kn, DIM, lane);

#pragma unroll
        for (int t = 0; t < 4; ++t) acc[t] = wmma_bf16(a1, b1[t], acc[t]);
    }

    const int half = lane >> 4, nlo = lane & 15;
#pragma unroll
    for (int t = 0; t < 4; ++t) {
#pragma unroll
        for (int j = 0; j < 8; ++j) {
            int m = row + j + 8 * half;
            int n = colb + t * 16 + nlo;
            Out[(size_t)m * DIM + n] = acc[t][j] + bo[n];
        }
    }
}

// ---------------------------------------------------------------------------
extern "C" void kernel_launch(void* const* d_in, const int* in_sizes, int n_in,
                              void* d_out, int out_size, void* d_ws, size_t ws_size,
                              hipStream_t stream) {
    (void)in_sizes; (void)n_in; (void)out_size; (void)ws_size;

    // setup_inputs order: key, query, value, Wq, Wk, Wv, Wo, bo
    const float* key   = (const float*)d_in[0];
    const float* query = (const float*)d_in[1];
    const float* value = (const float*)d_in[2];
    const float* Wq    = (const float*)d_in[3];
    const float* Wk    = (const float*)d_in[4];
    const float* Wv    = (const float*)d_in[5];
    const float* Wo    = (const float*)d_in[6];
    const float* bo    = (const float*)d_in[7];
    float* out = (float*)d_out;

    char* ws = (char*)d_ws;
    const size_t MB = 1024 * 1024;
    __bf16* Xqbf = (__bf16*)(ws);              //  8 MB  query bf16 (pre-scaled)
    __bf16* Xkbf = (__bf16*)(ws +  8 * MB);    //  8 MB  key   bf16
    __bf16* Xvbf = (__bf16*)(ws + 16 * MB);    //  8 MB  value bf16
    __bf16* Wqbf = (__bf16*)(ws + 24 * MB);    //  2 MB
    __bf16* Wkbf = (__bf16*)(ws + 26 * MB);    //  2 MB
    __bf16* Wvbf = (__bf16*)(ws + 28 * MB);    //  2 MB
    __bf16* Wobf = (__bf16*)(ws + 30 * MB);    //  2 MB
    __bf16* Qbf  = (__bf16*)(ws + 32 * MB);    //  8 MB  [B,H,S,64]
    __bf16* Kbf  = (__bf16*)(ws + 40 * MB);    //  8 MB  [B,H,S,64]
    __bf16* Vt   = (__bf16*)(ws + 48 * MB);    //  8 MB  [B,H,64,S]
    __bf16* Ctx  = (__bf16*)(ws + 56 * MB);    //  8 MB  [B,S,1024]

    const int M    = BATCH * SEQ;              // 4096
    const int NACT = M * DIM;                  // 4,194,304
    const int NW   = DIM * DIM;                // 1,048,576
    dim3 blk(256, 1, 1);

    // stage 0: conversions (1/sqrt(64)=0.125 folded into query)
    dim3 ga((NACT / 8 + 255) / 256);
    dim3 gw((NW / 8 + 255) / 256);
    hipLaunchKernelGGL(cvt_f32_bf16, ga, blk, 0, stream, query, Xqbf, NACT, 0.125f);
    hipLaunchKernelGGL(cvt_f32_bf16, ga, blk, 0, stream, key,   Xkbf, NACT, 1.0f);
    hipLaunchKernelGGL(cvt_f32_bf16, ga, blk, 0, stream, value, Xvbf, NACT, 1.0f);
    hipLaunchKernelGGL(cvt_f32_bf16, gw, blk, 0, stream, Wq,    Wqbf, NW,   1.0f);
    hipLaunchKernelGGL(cvt_f32_bf16, gw, blk, 0, stream, Wk,    Wkbf, NW,   1.0f);
    hipLaunchKernelGGL(cvt_f32_bf16, gw, blk, 0, stream, Wv,    Wvbf, NW,   1.0f);
    hipLaunchKernelGGL(cvt_f32_bf16, gw, blk, 0, stream, Wo,    Wobf, NW,   1.0f);

    // stage 1: QKV projections
    dim3 g1(M / 128, DIM / 64, 3);             // (32,16,3)
    hipLaunchKernelGGL(mha_proj_qkv, g1, blk, 0, stream,
                       Xqbf, Xkbf, Xvbf, Wqbf, Wkbf, Wvbf, Qbf, Kbf, Vt);

    // stage 2: attention (32 KB dynamic LDS: K/V double buffers + P tiles)
    dim3 g2(SEQ / 128, NHEAD, BATCH);          // (16,16,2)
    hipLaunchKernelGGL(mha_attn, g2, blk, 32768, stream, Qbf, Kbf, Vt, Ctx);

    // stage 3: output projection
    dim3 g3(M / 128, DIM / 64, 1);             // (32,16)
    hipLaunchKernelGGL(mha_oproj, g3, blk, 0, stream, Ctx, Wobf, bo, out);
}
